// LSTM_12283606467783
// MI455X (gfx1250) — matrix operand order repaired and checked by reading, not compiled, not run on hardware
//
#include <hip/hip_runtime.h>

#define B_TOT   4096
#define T_STEPS 256
#define HID     25
#define G4      100   // 4*HID
#define GPAD    112   // 7 tiles * 16
#define NT      7
#define ROWS    16    // batch rows (= 8 independent pairs) per block
#define BDIM    256   // 8 cooperating wave32s per chain

typedef _Float16 v16h __attribute__((ext_vector_type(16)));
typedef _Float16 v8h  __attribute__((ext_vector_type(8)));
typedef float    v8f  __attribute__((ext_vector_type(8)));

#if __has_builtin(__builtin_amdgcn_tanhf)
__device__ __forceinline__ float tanh_f(float x) { return __builtin_amdgcn_tanhf(x); }
#else
__device__ __forceinline__ float tanh_f(float x) {
  return 2.0f * __builtin_amdgcn_rcpf(1.0f + __expf(-2.0f * x)) - 1.0f;
}
#endif
__device__ __forceinline__ float sigm(float x) {
  return __builtin_amdgcn_rcpf(1.0f + __expf(-x));
}

// Gather A-matrix (16x32 f16, row-major in LDS) into a contiguous-k WMMA per-lane
// layout: lanes 0-15 hold K 0-15 of row M=lane, lanes 16-31 hold
// K 16-31 of row M=lane-16.
__device__ __forceinline__ v16h load_A(const _Float16* stage, int lane) {
  const _Float16* p = stage + (lane & 15) * 32 + ((lane & 16) ? 16 : 0);
  return *(const v16h*)(p);
}

// One 16x16 gate tile: acc = bias; acc += A_x * B_x; acc += A_h * B_h
__device__ __forceinline__ void gate_tile(int tile, int lane,
                                          v16h ax, v16h ah,
                                          const _Float16* BpL,  // [part][tile][lane][16]
                                          const float* bsumL,
                                          float* gbuf)          // [16][GPAD]
{
  const int nl = lane & 15;
  const int mb = (lane & 16) ? 8 : 0;
  float bv = bsumL[tile * 16 + nl];
  v8f acc;
#pragma unroll
  for (int j = 0; j < 8; ++j) acc[j] = bv;
  v16h bx = *(const v16h*)(BpL + (tile * 32 + lane) * 16);
  acc = __builtin_amdgcn_wmma_f32_16x16x32_f16(false, ax, false, bx,
                                               (short)0, acc, false, false);
  v16h bh = *(const v16h*)(BpL + ((NT + tile) * 32 + lane) * 16);
  acc = __builtin_amdgcn_wmma_f32_16x16x32_f16(false, ah, false, bh,
                                               (short)0, acc, false, false);
  const int n = tile * 16 + nl;
  if (n < G4) {
#pragma unroll
    for (int j = 0; j < 8; ++j) gbuf[(mb + j) * GPAD + n] = acc[j];
  }
}

__global__ __launch_bounds__(BDIM)
void lstm_conv_wmma(const float* __restrict__ xg,
                    const float* __restrict__ Wih0, const float* __restrict__ Whh0,
                    const float* __restrict__ bih0, const float* __restrict__ bhh0,
                    const float* __restrict__ Wih1, const float* __restrict__ Whh1,
                    const float* __restrict__ bih1, const float* __restrict__ bhh1,
                    const float* __restrict__ c1wg, const float* __restrict__ c1bg,
                    const float* __restrict__ c2wg, const float* __restrict__ c2bg,
                    float* __restrict__ out)
{
  __shared__ _Float16 Bpack[2 * 2 * NT * 32 * 16]; // [layer][x|h][tile][lane][16]
  __shared__ float    bsum[2 * GPAD];
  __shared__ float    cpad[2 * ROWS * 49];         // 7x7 zero-padded cell state
  __shared__ _Float16 stx[ROWS * 32];              // x(t) stage (K padded to 32)
  __shared__ _Float16 sth0[ROWS * 32];             // h0 stage (f16, written by ew0)
  __shared__ _Float16 sth1[ROWS * 32];             // h1 stage (f16, written by ew1)
  __shared__ float    gbuf[ROWS * GPAD];           // gate pre-activations
  __shared__ float    mid[2 * 8 * 5 * 25];         // conv1 output (relu'd)
  __shared__ float    c1w[90], c1b[5], c2w[90], c2b[2];

  const int tid  = threadIdx.x;
  const int lane = tid & 31;
  const int wave = tid >> 5;   // wave-uniform
  const int R0   = blockIdx.x * ROWS;

  // Two fixed per-thread (row, unit) slots cover ROWS*HID = 400 elements.
  const int  idx1v = tid + BDIM;
  const bool v1    = idx1v < ROWS * HID;           // slot 0 always valid (tid < 256)
  const int  r0 = tid / HID,                u0 = tid - r0 * HID;
  const int  r1 = v1 ? idx1v / HID : 0;
  const int  u1 = v1 ? idx1v - r1 * HID : 0;
  const int  p0 = (u0 / 5 + 1) * 7 + (u0 % 5) + 1; // padded 7x7 interior index
  const int  p1 = (u1 / 5 + 1) * 7 + (u1 % 5) + 1;
  const size_t off0 = (size_t)(R0 + r0) * T_STEPS * HID + u0; // x and out share indexing
  const size_t off1 = (size_t)(R0 + r1) * T_STEPS * HID + u1;

  // ---------------- one-time init ----------------
  for (int i = tid; i < 90; i += BDIM) { c1w[i] = c1wg[i]; c2w[i] = c2wg[i]; }
  if (tid < 5) c1b[tid] = c1bg[tid];
  if (tid < 2) c2b[tid] = c2bg[tid];
  for (int i = tid; i < 2 * ROWS * 49; i += BDIM) cpad[i] = 0.f;   // borders stay 0
  for (int i = tid; i < ROWS * 32; i += BDIM) {
    stx[i]  = (_Float16)0.f;  // K-padding cols >=25 never rewritten
    sth0[i] = (_Float16)0.f;  // also the h(t=0) = 0 initial state
    sth1[i] = (_Float16)0.f;
  }
  if (tid < GPAD) {
    bsum[tid]        = (tid < G4) ? (bih0[tid] + bhh0[tid]) : 0.f;
    bsum[GPAD + tid] = (tid < G4) ? (bih1[tid] + bhh1[tid]) : 0.f;
  }
  // B[k][n] = W[n][k]; lanes 0-15: col=lane, K 0-15; lanes 16-31: col=lane-16, K 16-31.
  for (int slot = tid; slot < 2 * 2 * NT * 32; slot += BDIM) {
    int lane_s = slot & 31; int q = slot >> 5;
    int tile = q % NT; q /= NT;
    int part = q & 1;  int layer = q >> 1;
    const float* W = (layer == 0) ? (part == 0 ? Wih0 : Whh0)
                                  : (part == 0 ? Wih1 : Whh1);
    int N  = tile * 16 + (lane_s & 15);
    int kb = (lane_s & 16) ? 16 : 0;
    _Float16* dst = &Bpack[slot * 16];
    for (int j = 0; j < 16; ++j) {
      int k = kb + j;
      float v = (N < G4 && k < HID) ? W[N * HID + k] : 0.f;
      dst[j] = (_Float16)v;
    }
  }
  __syncthreads();

  // prologue: stage x(t=0)
  stx[r0 * 32 + u0] = (_Float16)xg[off0];
  if (v1) stx[r1 * 32 + u1] = (_Float16)xg[off1];
  __syncthreads();

  // ---------------- 256-step recurrence, block-local, 6 barriers/step ----------------
  for (int t = 0; t < T_STEPS; ++t) {
    // Issue next step's input loads now; waited on only at the end of the step.
    float xr0 = 0.f, xr1 = 0.f;
    const bool have_next = (t + 1 < T_STEPS);
    if (have_next) {
      xr0 = xg[off0 + (size_t)(t + 1) * HID];
      if (v1) xr1 = xg[off1 + (size_t)(t + 1) * HID];
      if (t + 2 < T_STEPS)
        __builtin_prefetch(&xg[off0 + (size_t)(t + 2) * HID], 0, 0);
    }

#pragma unroll
    for (int layer = 0; layer < 2; ++layer) {
      // gates: one 16x16 N-tile per wave (wave-uniform guard -> EXEC all 1s)
      if (wave < NT) {
        const _Float16* aXs = (layer == 0) ? stx  : sth0;  // x(t) | new h0
        const _Float16* aHs = (layer == 0) ? sth0 : sth1;  // prev h0 | prev h1
        v16h ax = load_A(aXs, lane);
        v16h ah = load_A(aHs, lane);
        gate_tile(wave, lane, ax, ah,
                  Bpack + layer * (2 * NT * 32 * 16),
                  bsum + layer * GPAD, gbuf);
      }
      __syncthreads();

      // elementwise LSTM update; h is written straight into its consumer stage (f16)
      _Float16* hst = (layer == 0) ? sth0 : sth1;
      {
        float* cp = &cpad[(layer * ROWS + r0) * 49 + p0];
        const float* gb = &gbuf[r0 * GPAD];
        float c = sigm(gb[HID + u0]) * cp[0] + sigm(gb[u0]) * tanh_f(gb[2 * HID + u0]);
        float h = sigm(gb[3 * HID + u0]) * tanh_f(c);
        cp[0] = c;
        hst[r0 * 32 + u0] = (_Float16)h;
        if (layer == 1) out[off0 + (size_t)t * HID] = h;   // ys = h1 (pre-conv)
      }
      if (v1) {
        float* cp = &cpad[(layer * ROWS + r1) * 49 + p1];
        const float* gb = &gbuf[r1 * GPAD];
        float c = sigm(gb[HID + u1]) * cp[0] + sigm(gb[u1]) * tanh_f(gb[2 * HID + u1]);
        float h = sigm(gb[3 * HID + u1]) * tanh_f(c);
        cp[0] = c;
        hst[r1 * 32 + u1] = (_Float16)h;
        if (layer == 1) out[off1 + (size_t)t * HID] = h;
      }
      __syncthreads();
    }

    // conv1(2->5, 3x3, SAME) + relu: branch-free thanks to zero-padded cstate.
    // Flat-reshape pairing keeps samples (layer, s) on rows 2s, 2s+1 of this block.
    for (int idx = tid; idx < 2000; idx += BDIM) {
      int pos = idx % 25; int q = idx / 25;
      int oc = q % 5; q /= 5;
      int s = q & 7; int layer = q >> 3;
      int y = pos / 5, x = pos - y * 5;
      int base = y * 7 + x;
      float acc = c1b[oc];
#pragma unroll
      for (int ic = 0; ic < 2; ++ic) {
        const float* cin = &cpad[(layer * ROWS + 2 * s + ic) * 49];
        const float* w   = &c1w[(oc * 2 + ic) * 9];
#pragma unroll
        for (int ky = 0; ky < 3; ++ky)
#pragma unroll
          for (int kx = 0; kx < 3; ++kx)
            acc += cin[base + ky * 7 + kx] * w[ky * 3 + kx];
      }
      mid[((layer * 8 + s) * 5 + oc) * 25 + pos] = fmaxf(acc, 0.f);
    }
    __syncthreads();

    // conv2(5->2, 3x3, SAME) + sigmoid -> new padded cell states
    for (int idx = tid; idx < 800; idx += BDIM) {
      int pos = idx % 25; int q = idx / 25;
      int oc = q & 1; q >>= 1;
      int s = q & 7; int layer = q >> 3;
      int y = pos / 5, x = pos - y * 5;
      float acc = c2b[oc];
#pragma unroll
      for (int ic = 0; ic < 5; ++ic) {
        const float* m = &mid[((layer * 8 + s) * 5 + ic) * 25];
        const float* w = &c2w[(oc * 5 + ic) * 9];
#pragma unroll
        for (int dy = -1; dy <= 1; ++dy) {
          int yy = y + dy; if ((unsigned)yy >= 5u) continue;
#pragma unroll
          for (int dx = -1; dx <= 1; ++dx) {
            int xx = x + dx; if ((unsigned)xx >= 5u) continue;
            acc += m[yy * 5 + xx] * w[(dy + 1) * 3 + (dx + 1)];
          }
        }
      }
      cpad[(layer * ROWS + 2 * s + oc) * 49 + (y + 1) * 7 + x + 1] = sigm(acc);
    }

    // Retire the prefetched next-step inputs into the (now free) x stage,
    // then one barrier covers both conv2's cpad writes and these stores.
    if (have_next) {
      stx[r0 * 32 + u0] = (_Float16)xr0;
      if (v1) stx[r1 * 32 + u1] = (_Float16)xr1;
    }
    __syncthreads();
  }
}

extern "C" void kernel_launch(void* const* d_in, const int* in_sizes, int n_in,
                              void* d_out, int out_size, void* d_ws, size_t ws_size,
                              hipStream_t stream) {
  const float* xg   = (const float*)d_in[0];
  const float* Wih0 = (const float*)d_in[1];
  const float* Whh0 = (const float*)d_in[2];
  const float* bih0 = (const float*)d_in[3];
  const float* bhh0 = (const float*)d_in[4];
  const float* Wih1 = (const float*)d_in[5];
  const float* Whh1 = (const float*)d_in[6];
  const float* bih1 = (const float*)d_in[7];
  const float* bhh1 = (const float*)d_in[8];
  const float* c1w  = (const float*)d_in[9];
  const float* c1b  = (const float*)d_in[10];
  const float* c2w  = (const float*)d_in[11];
  const float* c2b  = (const float*)d_in[12];
  float* out = (float*)d_out;

  dim3 grid(B_TOT / ROWS);   // 256 independent recurrence chains (8 pairs each)
  dim3 block(BDIM);          // 8 wave32s cooperate on each chain
  lstm_conv_wmma<<<grid, block, 0, stream>>>(xg, Wih0, Whh0, bih0, bhh0,
                                             Wih1, Whh1, bih1, bhh1,
                                             c1w, c1b, c2w, c2b, out);
}
